// knrm_8727373545845
// MI455X (gfx1250) — matrix operand order repaired and checked by reading, not compiled, not run on hardware
//
#include <hip/hip_runtime.h>
#include <hip/hip_bf16.h>
#include <stdint.h>

#define B_   128
#define LQ   32
#define LD   1024
#define DIM  300
#define CC   128
#define NBK  11
#define PAD_ELEMS 2048   // zeroed tail pad for embedding buffers (bf16 elems)

typedef __attribute__((ext_vector_type(16))) __bf16 v16bf;
typedef __attribute__((ext_vector_type(8)))  float  v8f;

// mu / -1/(2*sigma^2)
__constant__ float c_mu[NBK]   = {1.0f,0.9f,0.7f,0.5f,0.3f,0.1f,-0.1f,-0.3f,-0.5f,-0.7f,-0.9f};
__constant__ float c_coef[NBK] = {-500000.0f,-50.0f,-50.0f,-50.0f,-50.0f,-50.0f,-50.0f,-50.0f,-50.0f,-50.0f,-50.0f};
// pair order from reference concat: (qu,du)(qu,dt)(qu,db)(qb,du)(qt,du)(qb,db)(qb,dt)(qt,db)(qt,dt)
__device__ const int c_qg[9] = {0,0,0,1,2,1,1,2,2};
__device__ const int c_dg[9] = {0,2,1,0,0,1,2,1,2};

union FragA { v16bf v; uint2 u2[4]; unsigned short h[16]; };
union FragB { v16bf v; uint4 u4[2]; unsigned short h[16]; };

__device__ __forceinline__ unsigned short f2bf(float f) {
  __bf16 h = (__bf16)f;
  return __builtin_bit_cast(unsigned short, h);
}

// A fragment: ISA 16-bit A(16x32) layout. chunks at +0 (e<8 -> K0+e) and +16 (e>=8 -> K0+8+e).
// Global alignment is 8B (row byte offset = 600*l), so use uint2 loads.
__device__ __forceinline__ void loadA(FragA& a, const unsigned short* p) {
  const uint2* p0 = (const uint2*)p;
  a.u2[0] = p0[0]; a.u2[1] = p0[1];
  const uint2* p1 = (const uint2*)(p + 16);
  a.u2[2] = p1[0]; a.u2[3] = p1[1];
}
// B fragment: same layout as A, chunks at +0 and +16 (16B-aligned) -> two b128 loads.
__device__ __forceinline__ void loadB(FragB& b, const unsigned short* p) {
  const uint4* q = (const uint4*)p;
  b.u4[0] = q[0]; b.u4[1] = q[2];
}

// ---------------- 1) embedding gather -> bf16 (block per token position) ----------------
__global__ void knrm_gather_bf16(const int* __restrict__ tok,
                                 const float* __restrict__ emb,
                                 unsigned short* __restrict__ out) {
  const int pos = blockIdx.x;
  const int t   = tok[pos];
  const float*    src = emb + (size_t)t * DIM;
  unsigned short* dst = out + (size_t)pos * DIM;
  for (int dd = threadIdx.x; dd < DIM; dd += blockDim.x)
    dst[dd] = f2bf(src[dd]);
}

__global__ void knrm_zero_u16(unsigned short* p, int n) {
  int i = blockIdx.x * blockDim.x + threadIdx.x;
  if (i < n) p[i] = 0;
}

// ---------------- 2) weight transpose/pad -> bf16 Wt[c][Kp] ----------------
__global__ void knrm_wprep(const float* __restrict__ W,
                           unsigned short* __restrict__ Wt, int K, int Kp) {
  int total = CC * Kp;
  for (int i = blockIdx.x * blockDim.x + threadIdx.x; i < total;
       i += gridDim.x * blockDim.x) {
    int c = i / Kp, kk = i - c * Kp;
    Wt[i] = (kk < K) ? f2bf(W[(long)kk * CC + c]) : (unsigned short)0;
  }
}

// ---------------- 3) n-gram conv GEMM + ReLU + eps + L2 norm ----------------
// grid: x = M-tiles (ceil(Lpos/16)), y = gram (0..2), z = batch. block = 256 (8 waves).
// Branch-free K loop: Wt is zero-padded in K, embedding buffers have a zeroed tail pad,
// and rows >= Lg are zero-filled at store time, so no guards are needed on A loads.
__global__ __launch_bounds__(256) void knrm_conv_norm(
    const unsigned short* __restrict__ embp,   // (B, Lpos, 300) bf16 (+ zeroed tail pad)
    const unsigned short* __restrict__ Wt0, const unsigned short* __restrict__ Wt1,
    const unsigned short* __restrict__ Wt2,
    const float* __restrict__ b0, const float* __restrict__ b1, const float* __restrict__ b2,
    unsigned short* __restrict__ out0, unsigned short* __restrict__ out1,
    unsigned short* __restrict__ out2,
    int Lpos)
{
  __shared__ float tile[16][CC];
  __shared__ float sinv[16];

  const int g  = blockIdx.y;
  const int bz = blockIdx.z;
  const unsigned short* Wt  = (g == 0) ? Wt0 : (g == 1) ? Wt1 : Wt2;
  const float*          bia = (g == 0) ? b0  : (g == 1) ? b1  : b2;
  unsigned short*       out = (g == 0) ? out0 : (g == 1) ? out1 : out2;

  const int Ksteps = (DIM * (g + 1) + 31) >> 5;  // 10/19/29
  const int Kp     = Ksteps << 5;                // 320/608/928
  const int Lg     = Lpos - g;
  const int Mbase  = blockIdx.x * 16;

  const int tid  = threadIdx.x;
  const int wave = tid >> 5, lane = tid & 31;
  const int mrow = lane & 15;
  const int l    = Mbase + mrow;
  const int baseKa = (lane >= 16) ? 8  : 0;
  const int baseKb = (lane >= 16) ? 8  : 0;
  const int ncol   = (wave << 4) + mrow;         // channel column 0..127

  const unsigned short* Arow = embp + ((size_t)bz * Lpos + l) * DIM + baseKa;
  const unsigned short* Brow = Wt + (size_t)ncol * Kp + baseKb;

  v8f acc = {};
  FragA a0, a1;
  FragB bf0, bf1;
  loadA(a0,  Arow);
  loadB(bf0, Brow);
  for (int kt = 0; kt + 1 < Ksteps; ++kt) {
    const int K0 = (kt + 1) << 5;
    loadA(a1,  Arow + K0);
    loadB(bf1, Brow + K0);
    acc = __builtin_amdgcn_wmma_f32_16x16x32_bf16(false, a0.v, false, bf0.v,
                                                  (short)0, acc, false, false);
    a0 = a1; bf0 = bf1;
  }
  acc = __builtin_amdgcn_wmma_f32_16x16x32_bf16(false, a0.v, false, bf0.v,
                                                (short)0, acc, false, false);

  // bias + relu + eps, stage tile in LDS: C/D layout M = r + 8*(lane>=16), N = ncol
  const float bc    = bia[ncol];
  const int   rbase = (lane >= 16) ? 8 : 0;
#pragma unroll
  for (int r = 0; r < 8; ++r) {
    float v = fmaxf(acc[r] + bc, 0.0f) + 1e-9f;
    tile[rbase + r][ncol] = v;
  }
  __syncthreads();

  if (tid < 16) {
    float s = 0.0f;
#pragma unroll 4
    for (int c = 0; c < CC; ++c) { float v = tile[tid][c]; s += v * v; }
    sinv[tid] = 1.0f / fmaxf(sqrtf(s), 1e-10f);
  }
  __syncthreads();

#pragma unroll
  for (int e = 0; e < 8; ++e) {
    int idx = tid * 8 + e;                 // 0..2047
    int row = idx >> 7, col = idx & 127;
    int lr  = Mbase + row;
    if (lr < Lpos) {
      float v = (lr < Lg) ? tile[row][col] * sinv[row] : 0.0f;  // zero-pad rows >= Lg
      out[((size_t)bz * Lpos + lr) * CC + col] = f2bf(v);
    }
  }
}

// ---------------- 4) sim GEMM + Gaussian kernel pooling ----------------
// grid: x = q M-tiles (2), y = pair (0..8), z = batch. block = 256 (8 waves).
// Pooling uses all 256 threads: thread = (row 0..15, colgroup 0..15), 11 register accums.
__global__ __launch_bounds__(256) void knrm_pool(
    const unsigned short* __restrict__ qn0, const unsigned short* __restrict__ qn1,
    const unsigned short* __restrict__ qn2,
    const unsigned short* __restrict__ dn0, const unsigned short* __restrict__ dn1,
    const unsigned short* __restrict__ dn2,
    const float* __restrict__ qwm, const float* __restrict__ dwm,
    float* __restrict__ feats)
{
  __shared__ float simt[16][CC];
  __shared__ float mdv[CC];
  __shared__ float red2[16][16][NBK];   // [row][colgroup][nb]
  __shared__ float red[16][NBK];

  const int pair = blockIdx.y, bz = blockIdx.z;
  const int qg = c_qg[pair], dg = c_dg[pair];
  const unsigned short* qn = (qg == 0) ? qn0 : (qg == 1) ? qn1 : qn2;
  const unsigned short* dn = (dg == 0) ? dn0 : (dg == 1) ? dn1 : dn2;
  const int Lqg = LQ - qg, Ldg = LD - dg;
  const int Mbase = blockIdx.x * 16;

  const int tid  = threadIdx.x;
  const int wave = tid >> 5, lane = tid & 31;
  const int mrow = lane & 15;
  const int baseKa = (lane >= 16) ? 8  : 0;
  const int baseKb = (lane >= 16) ? 8  : 0;

  // preload A fragments for the q tile (rows padded with zeros in ws, always in-bounds)
  const unsigned short* qrow = qn + ((size_t)bz * LQ + (Mbase + mrow)) * CC + baseKa;
  FragB a[4];
#pragma unroll
  for (int kt = 0; kt < 4; ++kt) {
    const int K0 = kt << 5;
    a[kt].u4[0] = *(const uint4*)(qrow + K0);
    a[kt].u4[1] = *(const uint4*)(qrow + K0 + 16);
  }

  const int row_t = tid >> 4;     // 0..15
  const int cg_t  = tid & 15;     // column group: columns cg*8 .. cg*8+7
  float ps[NBK];
#pragma unroll
  for (int nb = 0; nb < NBK; ++nb) ps[nb] = 0.0f;

  for (int j = 0; j < 8; ++j) {
    const int ncol = j * CC + (wave << 4) + mrow;               // d position
    const unsigned short* drow = dn + ((size_t)bz * LD + ncol) * CC;
    if (j < 7) __builtin_prefetch(drow + CC * CC, 0, 1);        // next block of d rows

    v8f acc = {};
#pragma unroll
    for (int kt = 0; kt < 4; ++kt) {
      FragB b;
      const uint4* pb = (const uint4*)(drow + (kt << 5) + baseKb);
      b.u4[0] = pb[0]; b.u4[1] = pb[2];
      acc = __builtin_amdgcn_wmma_f32_16x16x32_bf16(false, a[kt].v, false, b.v,
                                                    (short)0, acc, false, false);
    }

    const int rbase = (lane >= 16) ? 8 : 0;
    const int scol  = (wave << 4) + mrow;
#pragma unroll
    for (int r = 0; r < 8; ++r) simt[rbase + r][scol] = acc[r];

    if (tid < CC) {
      int n = j * CC + tid;
      mdv[tid] = (n < Ldg) ? dwm[(size_t)bz * LD + n] : 0.0f;
    }
    __syncthreads();

    // all 256 threads: 8 columns x 11 kernels each
#pragma unroll 2
    for (int cc = 0; cc < 8; ++cc) {
      const int c = (cg_t << 3) + cc;
      const float s  = simt[row_t][c];
      const float md = mdv[c];
#pragma unroll
      for (int nb = 0; nb < NBK; ++nb) {
        float dmu = s - c_mu[nb];
        ps[nb] += __expf(dmu * dmu * c_coef[nb]) * md;
      }
    }
    __syncthreads();
  }

#pragma unroll
  for (int nb = 0; nb < NBK; ++nb) red2[row_t][cg_t][nb] = ps[nb];
  __syncthreads();

  if (tid < 16 * NBK) {
    const int row = tid / NBK, nb = tid % NBK;
    float s = 0.0f;
#pragma unroll
    for (int cg = 0; cg < 16; ++cg) s += red2[row][cg][nb];
    int   lq = Mbase + row;
    float mq = (lq < Lqg) ? qwm[(size_t)bz * LQ + lq] : 0.0f;
    red[row][nb] = __logf(fmaxf(s, 1e-10f)) * 0.01f * mq;
  }
  __syncthreads();

  if (tid < NBK) {
    float s = 0.0f;
#pragma unroll
    for (int r = 0; r < 16; ++r) s += red[r][tid];
    atomicAdd(&feats[(size_t)bz * (9 * NBK) + pair * NBK + tid], s);
  }
}

// ---------------- 5) misc ----------------
__global__ void knrm_zero_f32(float* p, int n) {
  int i = blockIdx.x * blockDim.x + threadIdx.x;
  if (i < n) p[i] = 0.0f;
}

__global__ void knrm_final(const float* __restrict__ feats,
                           const float* __restrict__ w,
                           const float* __restrict__ b,
                           float* __restrict__ out) {
  int bi = threadIdx.x;
  if (bi < B_) {
    float s = b[0];
#pragma unroll 3
    for (int i = 0; i < 9 * NBK; ++i) s += feats[bi * 9 * NBK + i] * w[i];
    out[bi] = tanhf(s);
  }
}

// ---------------- launch ----------------
extern "C" void kernel_launch(void* const* d_in, const int* in_sizes, int n_in,
                              void* d_out, int out_size, void* d_ws, size_t ws_size,
                              hipStream_t stream) {
  const int*   qwt     = (const int*)d_in[0];
  const int*   dwt     = (const int*)d_in[1];
  const float* qwm     = (const float*)d_in[2];
  const float* dwm     = (const float*)d_in[3];
  const float* emb     = (const float*)d_in[4];
  const float* W_uni   = (const float*)d_in[5];
  const float* b_uni   = (const float*)d_in[6];
  const float* W_bi    = (const float*)d_in[7];
  const float* b_bi    = (const float*)d_in[8];
  const float* W_tri   = (const float*)d_in[9];
  const float* b_tri   = (const float*)d_in[10];
  const float* dense_w = (const float*)d_in[11];
  const float* dense_b = (const float*)d_in[12];
  float*       out     = (float*)d_out;

  char*  ws  = (char*)d_ws;
  size_t off = 0;
  auto alloc = [&](size_t bytes) -> char* {
    char* p = ws + off;
    off = (off + bytes + 255) & ~(size_t)255;
    return p;
  };
  unsigned short* qe  = (unsigned short*)alloc(((size_t)B_ * LQ * DIM + PAD_ELEMS) * 2);
  unsigned short* de  = (unsigned short*)alloc(((size_t)B_ * LD * DIM + PAD_ELEMS) * 2);
  unsigned short* Wt0 = (unsigned short*)alloc((size_t)CC * 320 * 2);
  unsigned short* Wt1 = (unsigned short*)alloc((size_t)CC * 608 * 2);
  unsigned short* Wt2 = (unsigned short*)alloc((size_t)CC * 928 * 2);
  unsigned short* qn0 = (unsigned short*)alloc((size_t)B_ * LQ * CC * 2);
  unsigned short* qn1 = (unsigned short*)alloc((size_t)B_ * LQ * CC * 2);
  unsigned short* qn2 = (unsigned short*)alloc((size_t)B_ * LQ * CC * 2);
  unsigned short* dn0 = (unsigned short*)alloc((size_t)B_ * LD * CC * 2);
  unsigned short* dn1 = (unsigned short*)alloc((size_t)B_ * LD * CC * 2);
  unsigned short* dn2 = (unsigned short*)alloc((size_t)B_ * LD * CC * 2);
  float*          fts = (float*)alloc((size_t)B_ * 9 * NBK * 4);

  // 0) zero feature accumulator + embedding tail pads (keeps K-tail loads finite/zero)
  knrm_zero_f32<<<(B_ * 9 * NBK + 255) / 256, 256, 0, stream>>>(fts, B_ * 9 * NBK);
  knrm_zero_u16<<<(PAD_ELEMS + 255) / 256, 256, 0, stream>>>(qe + (size_t)B_ * LQ * DIM, PAD_ELEMS);
  knrm_zero_u16<<<(PAD_ELEMS + 255) / 256, 256, 0, stream>>>(de + (size_t)B_ * LD * DIM, PAD_ELEMS);

  // 1) gathers (f32 -> bf16), one block per token position
  knrm_gather_bf16<<<B_ * LQ, 256, 0, stream>>>(qwt, emb, qe);
  knrm_gather_bf16<<<B_ * LD, 256, 0, stream>>>(dwt, emb, de);

  // 2) weight prep (transpose + pad to K multiples of 32)
  knrm_wprep<<<(CC * 320 + 255) / 256, 256, 0, stream>>>(W_uni, Wt0, 300, 320);
  knrm_wprep<<<(CC * 608 + 255) / 256, 256, 0, stream>>>(W_bi,  Wt1, 600, 608);
  knrm_wprep<<<(CC * 928 + 255) / 256, 256, 0, stream>>>(W_tri, Wt2, 900, 928);

  // 3) conv + norm: docs (64 M-tiles) and queries (2 M-tiles), 3 grams x 128 batches
  dim3 gd(64, 3, B_), gq(2, 3, B_);
  knrm_conv_norm<<<gd, 256, 0, stream>>>(de, Wt0, Wt1, Wt2, b_uni, b_bi, b_tri,
                                         dn0, dn1, dn2, LD);
  knrm_conv_norm<<<gq, 256, 0, stream>>>(qe, Wt0, Wt1, Wt2, b_uni, b_bi, b_tri,
                                         qn0, qn1, qn2, LQ);

  // 4) sim + kernel pooling: 2 q-tiles x 9 pairs x 128 batches
  dim3 gp(2, 9, B_);
  knrm_pool<<<gp, 256, 0, stream>>>(qn0, qn1, qn2, dn0, dn1, dn2, qwm, dwm, fts);

  // 5) dense + tanh
  knrm_final<<<1, 128, 0, stream>>>(fts, dense_w, dense_b, out);
}
